// GCNDelta_10771777979153
// MI455X (gfx1250) — matrix operand order repaired and checked by reading, not compiled, not run on hardware
//
#include <hip/hip_runtime.h>
#include <stdint.h>

// ---------------------------------------------------------------------------
// Types for CDNA5 WMMA (wave32, 16x16x32 bf16 -> f32)
// ---------------------------------------------------------------------------
typedef __attribute__((ext_vector_type(16))) __bf16 v16bf;
typedef __attribute__((ext_vector_type(8)))  float  v8f;
typedef unsigned int u32x4 __attribute__((ext_vector_type(4)));

union Frag {            // 32 bytes: one A (16x32) or B (32x16) bf16 fragment per lane
    u32x4 u[2];
    v16bf v;
};

__device__ __forceinline__ uint16_t f2bf(float f) {
    unsigned int u = __float_as_uint(f);
    u += 0x7FFFu + ((u >> 16) & 1u);   // round-to-nearest-even
    return (uint16_t)(u >> 16);
}

static constexpr int BATCH = 16;
static constexpr int NPTS  = 2048;
static constexpr int KNN   = 16;
static constexpr int FDIM  = 254;
static constexpr int DIN   = 256;   // 2 + 254
static constexpr int HDIM  = 512;
static constexpr int MROWS = BATCH * NPTS;   // 32768

// ---------------------------------------------------------------------------
// kNN: one thread per query point; batch positions staged in LDS.
// Fully unrolled 16-slot insertion keeps everything in registers.
// ---------------------------------------------------------------------------
__global__ __launch_bounds__(256) void knn_kernel(const float* __restrict__ pos,
                                                  int* __restrict__ idx) {
    __shared__ float sx[NPTS];
    __shared__ float sy[NPTS];
    const int b     = blockIdx.x >> 3;       // 8 chunks of 256 queries per batch
    const int chunk = blockIdx.x & 7;
    const float* pb = pos + (size_t)b * NPTS * 2;
    for (int j = threadIdx.x; j < NPTS; j += 256) {
        sx[j] = pb[2 * j];
        sy[j] = pb[2 * j + 1];
    }
    __syncthreads();

    const int q = chunk * 256 + threadIdx.x;
    const float qx = sx[q], qy = sy[q];

    float bd[KNN];
    int   bi[KNN];
#pragma unroll
    for (int t = 0; t < KNN; ++t) { bd[t] = 3.0e38f; bi[t] = 0; }

    for (int j = 0; j < NPTS; ++j) {
        const float dx = sx[j] - qx;
        const float dy = sy[j] - qy;
        const float d2 = dx * dx + dy * dy;
        if (d2 < bd[KNN - 1]) {
#pragma unroll
            for (int p = KNN - 1; p >= 1; --p) {
                if (d2 < bd[p - 1])      { bd[p] = bd[p - 1]; bi[p] = bi[p - 1]; }
                else if (d2 < bd[p])     { bd[p] = d2;        bi[p] = j;         }
            }
            if (d2 < bd[0]) { bd[0] = d2; bi[0] = j; }
        }
    }
    int* op = idx + ((size_t)b * NPTS + q) * KNN;
#pragma unroll
    for (int t = 0; t < KNN; ++t) op[t] = bi[t];
}

// ---------------------------------------------------------------------------
// Weight prep: W [K,N] f32 -> Wt [N,K] bf16 (row-major transposed) so B
// fragments become contiguous 16-byte loads per lane.
// ---------------------------------------------------------------------------
__global__ void wt_kernel(const float* __restrict__ W, uint16_t* __restrict__ Wt,
                          int K, int Nn) {
    const int i = blockIdx.x * blockDim.x + threadIdx.x;
    if (i < K * Nn) {
        const int n = i / K;
        const int k = i - n * K;
        Wt[i] = f2bf(W[k * Nn + n]);
    }
}

// ---------------------------------------------------------------------------
// Input aggregation: x0[node,c] = bf16( mean_j concat(pos,feat)[b, idx[j], c] )
// One block per node, one thread per channel; per-neighbor reads are coalesced.
// ---------------------------------------------------------------------------
__global__ __launch_bounds__(DIN) void gather_in_kernel(const float* __restrict__ pos,
                                                        const float* __restrict__ feat,
                                                        const int* __restrict__ idx,
                                                        uint16_t* __restrict__ x0) {
    const int node = blockIdx.x;
    const int b    = node >> 11;
    const int c    = threadIdx.x;
    const int* ip  = idx + (size_t)node * KNN;
    float acc = 0.0f;
#pragma unroll
    for (int j = 0; j < KNN; ++j) {
        const int g = b * NPTS + ip[j];
        acc += (c < 2) ? pos[(size_t)g * 2 + c]
                       : feat[(size_t)g * FDIM + (c - 2)];
    }
    x0[(size_t)node * DIN + c] = f2bf(acc * 0.0625f);
}

// ---------------------------------------------------------------------------
// WMMA GEMM:  C[M,N] = A[M,K](bf16) * Bt[N,K](bf16)^T  (+bias)(+relu)
//   block = 128 threads = 4 waves (2x2); block tile 64x128, wave tile 32x64.
//   EPI: 0 = bias+relu -> bf16 ; 1 = bias -> bf16 ; 2 = none -> f32
// Fragment layouts follow CDNA5 ISA 7.12.2:
//   A lane(half,m): VGPR0-3 = K[half*8 .. +7], VGPR4-7 = K[16+half*8 .. +7]
//   B lane(half,n): VGPR0-3 = K[half*8 .. +7], VGPR4-7 = K[16+half*8 .. +7]  (from Bt row n)
//   C lane(half,n): VGPR r  = row half*8 + r, col n
// ---------------------------------------------------------------------------
template <int KDIM, int EPI>
__global__ __launch_bounds__(128) void gemm_wmma(const uint16_t* __restrict__ A,
                                                 const uint16_t* __restrict__ Bt,
                                                 const float* __restrict__ bias,
                                                 uint16_t* __restrict__ obf,
                                                 float* __restrict__ of32,
                                                 int Nn) {
    const int tid  = threadIdx.x;
    const int lane = tid & 31;
    const int w    = tid >> 5;
    const int wm   = w >> 1, wn = w & 1;
    const int half = lane >> 4, lm = lane & 15;
    const int m0   = blockIdx.y * 64 + wm * 32;
    const int n0   = blockIdx.x * 128 + wn * 64;

    v8f acc[2][4] = {};

    const uint16_t* arow0 = A + (size_t)(m0 + lm) * KDIM;
    const uint16_t* arow1 = arow0 + (size_t)16 * KDIM;
    const uint16_t* brow[4];
#pragma unroll
    for (int t = 0; t < 4; ++t)
        brow[t] = Bt + (size_t)(n0 + 16 * t + lm) * KDIM;

    for (int k = 0; k < KDIM; k += 32) {
        Frag a0, a1;
        a0.u[0] = *(const u32x4*)(arow0 + k + half * 8);
        a0.u[1] = *(const u32x4*)(arow0 + k + 16 + half * 8);
        a1.u[0] = *(const u32x4*)(arow1 + k + half * 8);
        a1.u[1] = *(const u32x4*)(arow1 + k + 16 + half * 8);
        Frag bf[4];
#pragma unroll
        for (int t = 0; t < 4; ++t) {
            bf[t].u[0] = *(const u32x4*)(brow[t] + k + half * 8);
            bf[t].u[1] = *(const u32x4*)(brow[t] + k + 16 + half * 8);
        }
#pragma unroll
        for (int t = 0; t < 4; ++t) {
            acc[0][t] = __builtin_amdgcn_wmma_f32_16x16x32_bf16(
                false, a0.v, false, bf[t].v, (short)0, acc[0][t], false, false);
            acc[1][t] = __builtin_amdgcn_wmma_f32_16x16x32_bf16(
                false, a1.v, false, bf[t].v, (short)0, acc[1][t], false, false);
        }
    }

#pragma unroll
    for (int i = 0; i < 2; ++i) {
#pragma unroll
        for (int t = 0; t < 4; ++t) {
            const int col = n0 + 16 * t + lm;
            const float bv = (EPI == 2) ? 0.0f : bias[col];
            const int rowbase = m0 + 16 * i + half * 8;
#pragma unroll
            for (int r = 0; r < 8; ++r) {
                float v = acc[i][t][r] + bv;
                if (EPI == 0) v = v > 0.0f ? v : 0.0f;
                if (EPI == 2)
                    of32[(size_t)(rowbase + r) * Nn + col] = v;
                else
                    obf[(size_t)(rowbase + r) * Nn + col] = f2bf(v);
            }
        }
    }
}

// ---------------------------------------------------------------------------
// Final: neighbor-mean of y, + b2, tanh scaling, split into the two outputs.
// ---------------------------------------------------------------------------
__global__ __launch_bounds__(DIN) void final_kernel(const float* __restrict__ y,
                                                    const int* __restrict__ idx,
                                                    const float* __restrict__ b2,
                                                    float* __restrict__ out) {
    const int node = blockIdx.x;
    const int b    = node >> 11;
    const int c    = threadIdx.x;
    const int* ip  = idx + (size_t)node * KNN;
    float acc = 0.0f;
#pragma unroll
    for (int j = 0; j < KNN; ++j) {
        const int g = b * NPTS + ip[j];
        acc += y[(size_t)g * DIN + c];
    }
    const float v = acc * 0.0625f + b2[c];
    float* outPos  = out;
    float* outFeat = out + (size_t)MROWS * 2;
    if (c < 2)
        outPos[(size_t)node * 2 + c] = 0.4f * tanhf(v);
    else
        outFeat[(size_t)node * FDIM + (c - 2)] = 0.04f * tanhf(v);
}

// ---------------------------------------------------------------------------
// Launch
// ---------------------------------------------------------------------------
extern "C" void kernel_launch(void* const* d_in, const int* in_sizes, int n_in,
                              void* d_out, int out_size, void* d_ws, size_t ws_size,
                              hipStream_t stream) {
    const float* pos  = (const float*)d_in[0];
    const float* feat = (const float*)d_in[1];
    const float* W1   = (const float*)d_in[2];
    const float* b1   = (const float*)d_in[3];
    const float* Wm1  = (const float*)d_in[4];
    const float* bm1  = (const float*)d_in[5];
    const float* Wm2  = (const float*)d_in[6];
    const float* bm2  = (const float*)d_in[7];
    const float* W2   = (const float*)d_in[8];
    const float* b2   = (const float*)d_in[9];
    float* out = (float*)d_out;

    char* base = (char*)d_ws;
    size_t off = 0;
    auto take = [&](size_t bytes) -> char* {
        char* p = base + off;
        off = (off + bytes + 255) & ~(size_t)255;
        return p;
    };
    int*      idx_ws = (int*)     take((size_t)MROWS * KNN * 4);
    uint16_t* Wt1    = (uint16_t*)take((size_t)HDIM * DIN * 2);
    uint16_t* Wtm1   = (uint16_t*)take((size_t)HDIM * HDIM * 2);
    uint16_t* Wtm2   = (uint16_t*)take((size_t)HDIM * HDIM * 2);
    uint16_t* Wt2    = (uint16_t*)take((size_t)DIN * HDIM * 2);
    uint16_t* x0     = (uint16_t*)take((size_t)MROWS * DIN * 2);
    uint16_t* h1     = (uint16_t*)take((size_t)MROWS * HDIM * 2);   // also h3
    uint16_t* h2     = (uint16_t*)take((size_t)MROWS * HDIM * 2);
    float*    ybuf   = (float*)   take((size_t)MROWS * DIN * 4);

    // 1. kNN indices
    knn_kernel<<<dim3(BATCH * 8), dim3(256), 0, stream>>>(pos, idx_ws);

    // 2. weight transpose + bf16 convert
    wt_kernel<<<dim3((DIN * HDIM + 255) / 256), dim3(256), 0, stream>>>(W1,  Wt1,  DIN,  HDIM);
    wt_kernel<<<dim3((HDIM * HDIM + 255) / 256), dim3(256), 0, stream>>>(Wm1, Wtm1, HDIM, HDIM);
    wt_kernel<<<dim3((HDIM * HDIM + 255) / 256), dim3(256), 0, stream>>>(Wm2, Wtm2, HDIM, HDIM);
    wt_kernel<<<dim3((HDIM * DIN + 255) / 256), dim3(256), 0, stream>>>(W2,  Wt2,  HDIM, DIN);

    // 3. neighbor-mean of concat(pos,features) -> bf16
    gather_in_kernel<<<dim3(MROWS), dim3(DIN), 0, stream>>>(pos, feat, idx_ws, x0);

    // 4. GEMM chain (M=32768; block tile 64x128)
    gemm_wmma<DIN, 0><<<dim3(HDIM / 128, MROWS / 64), dim3(128), 0, stream>>>(
        x0, Wt1, b1, h1, nullptr, HDIM);                       // relu(x0@W1+b1)
    gemm_wmma<HDIM, 0><<<dim3(HDIM / 128, MROWS / 64), dim3(128), 0, stream>>>(
        h1, Wtm1, bm1, h2, nullptr, HDIM);                     // relu(h1@Wm1+bm1)
    gemm_wmma<HDIM, 1><<<dim3(HDIM / 128, MROWS / 64), dim3(128), 0, stream>>>(
        h2, Wtm2, bm2, h1, nullptr, HDIM);                     // h2@Wm2+bm2 -> h3(=h1)
    gemm_wmma<HDIM, 2><<<dim3(DIN / 128, MROWS / 64), dim3(128), 0, stream>>>(
        h1, Wt2, nullptr, nullptr, ybuf, DIN);                 // h3@W2 -> f32

    // 5. neighbor-mean + b2 + tanh scaling -> outputs
    final_kernel<<<dim3(MROWS), dim3(DIN), 0, stream>>>(ybuf, idx_ws, b2, out);
}
